// Decoder_83348135346455
// MI455X (gfx1250) — matrix operand order repaired and checked by reading, not compiled, not run on hardware
//
#include <hip/hip_runtime.h>
#include <hip/hip_bf16.h>

typedef __attribute__((ext_vector_type(16))) __bf16 v16bf;
typedef __attribute__((ext_vector_type(8)))  float  v8f;

#define DEC_B   4096
#define DEC_T   128
#define DEC_H   128

// Hardware-transcendental helpers (v_exp_f32 / v_rcp_f32 instead of ocml polys)
__device__ __forceinline__ float fast_tanh(float x) {
    float e = __expf(2.0f * x);
    return 1.0f - 2.0f * __builtin_amdgcn_rcpf(e + 1.0f);
}
__device__ __forceinline__ float fast_sig(float x) {
    return __builtin_amdgcn_rcpf(1.0f + __expf(-x));
}

__device__ __forceinline__ v8f wmma_bf16(const v16bf& a, const v16bf& b, const v8f& c) {
    return __builtin_amdgcn_wmma_f32_16x16x32_bf16(false, a, false, b, (short)0, c, false, false);
}

// Load an A-fragment (16x32 bf16) for this lane from a bf16 LDS row.
// Per ISA layout: element i<8 -> k = ks*32 + aOff + i ; i>=8 -> +8 more.
// Both 8-element chunks are 16B-aligned contiguous runs -> ds_load_b128 pairs.
__device__ __forceinline__ v16bf load_a_frag(const __bf16* rowPtr, int ks, int aOff) {
    const uint4* rp = (const uint4*)rowPtr;
    int c0 = (ks * 32 + aOff) >> 3;            // in 8-bf16 (16B) units
    union { uint4 q[2]; v16bf v; } u;
    u.q[0] = rp[c0];
    u.q[1] = rp[c0 + 2];                       // +16 bf16 elements
    return u.v;
}

// ---------------------------------------------------------------------------
// Kernel 0: pack weight matrices into WMMA B-fragment layout (bf16).
//  B-fragment convention (16x16x32): element i of lane L holds
//    n = ntile*16 + (L&15),  k = kstep*32 + (L>=16 ? 8 : 0) + i + (i>=8 ? 8 : 0)
//  Storage: frag-major, ((frag*32 + lane)*16 + i) bf16 elements.
// Regions:
//  frag   0..63  : [W1_h | W1_c]^T   (8 ntiles x 8 ksteps, K=256) = attn_W1[n][k], k<256
//  frag  64..191 : W_hh^T            (32 ntiles x 4 ksteps, K=128) = W_hh[n][k]
//  frag 192..223 : W1_e^T            (8 ntiles x 4 ksteps, K=128) = attn_W1[n][256+k]
// ---------------------------------------------------------------------------
__global__ void __launch_bounds__(32) prep_frags_kernel(
    const float* __restrict__ attn_W1,   // (128, 384)
    const float* __restrict__ W_hh,      // (512, 128)
    __bf16* __restrict__ w1hc_f, __bf16* __restrict__ whh_f, __bf16* __restrict__ w1e_f)
{
    int frag = blockIdx.x;
    int lane = threadIdx.x;
    int nLocal = lane & 15;
    int kOff   = (lane >= 16) ? 8 : 0;

    if (frag < 64) {
        int nt = frag >> 3, ks = frag & 7;
        int n = nt * 16 + nLocal;
        #pragma unroll
        for (int i = 0; i < 16; ++i) {
            int k = ks * 32 + kOff + i + (i >= 8 ? 8 : 0);    // 0..255
            w1hc_f[(size_t)frag * 512 + lane * 16 + i] = (__bf16)attn_W1[n * 384 + k];
        }
    } else if (frag < 192) {
        int f2 = frag - 64;
        int nt = f2 >> 2, ks = f2 & 3;
        int n = nt * 16 + nLocal;                             // 0..511
        #pragma unroll
        for (int i = 0; i < 16; ++i) {
            int k = ks * 32 + kOff + i + (i >= 8 ? 8 : 0);    // 0..127
            whh_f[(size_t)f2 * 512 + lane * 16 + i] = (__bf16)W_hh[n * 128 + k];
        }
    } else {
        int f2 = frag - 192;
        int nt = f2 >> 2, ks = f2 & 3;
        int n = nt * 16 + nLocal;
        #pragma unroll
        for (int i = 0; i < 16; ++i) {
            int k = ks * 32 + kOff + i + (i >= 8 ? 8 : 0);
            w1e_f[(size_t)f2 * 512 + lane * 16 + i] = (__bf16)attn_W1[n * 384 + 256 + k];
        }
    }
}

// ---------------------------------------------------------------------------
// Kernel 1: enc_proj = input_encoded @ W1_e^T  (bf16 WMMA GEMM, 524288x128x128)
// Also emits a bf16 copy of input_encoded (halves the per-step streaming
// traffic of the scan kernel; 2x134MB mostly fits the 192MB L2).
// ---------------------------------------------------------------------------
__global__ void __launch_bounds__(128) enc_proj_kernel(
    const float* __restrict__ X,            // (B*T, 128) fp32
    const __bf16* __restrict__ w1e_f,       // packed B fragments
    __bf16* __restrict__ encB,              // (B*T, 128) bf16 out
    __bf16* __restrict__ encProjB)          // (B*T, 128) bf16 out
{
    __shared__ __bf16 Xb[64][128];
    int row0 = blockIdx.x * 64;

    for (int idx = threadIdx.x; idx < 64 * 128; idx += 128) {
        int r = idx >> 7, cc = idx & 127;
        __bf16 bv = (__bf16)X[(size_t)(row0 + r) * 128 + cc];
        Xb[r][cc] = bv;
        encB[(size_t)(row0 + r) * 128 + cc] = bv;
    }
    __syncthreads();

    int wave = threadIdx.x >> 5, lane = threadIdx.x & 31;
    int m0 = wave * 16;
    int mA = lane & 15;
    int aOff = (lane >= 16) ? 8 : 0;
    int nLocal = lane & 15;
    int mBase = (lane >= 16) ? 8 : 0;
    const v16bf* w1e_v = (const v16bf*)w1e_f;

    #pragma unroll
    for (int nt = 0; nt < 8; ++nt) {
        v8f acc = {};
        #pragma unroll
        for (int ks = 0; ks < 4; ++ks) {
            v16bf a = load_a_frag(&Xb[m0 + mA][0], ks, aOff);
            v16bf b = w1e_v[(nt * 4 + ks) * 32 + lane];
            acc = wmma_bf16(a, b, acc);
        }
        int n = nt * 16 + nLocal;
        #pragma unroll
        for (int r = 0; r < 8; ++r) {
            int m = r + mBase;
            encProjB[(size_t)(row0 + m0 + m) * 128 + n] = (__bf16)acc[r];
        }
    }
}

// ---------------------------------------------------------------------------
// Kernel 2: persistent scan. One block = 16 batch rows, all 127 time steps.
// [h|c] lives in LDS as a bf16 16x256 tile (direct WMMA A-side source via
// ds_load_b128); cell state kept fp32 for the recurrence. Per step:
//   hc_proj = [h|c] @ [W1_h|W1_c]^T + b1          (WMMA, M16 N128 K256)
//   e[t'] = w2 . tanh(enc_proj + hc_proj) + b2    (VALU, bf16 streamed)
//   beta = softmax(e); context = beta @ enc_tile  (VALU, bf16 streamed)
//   y~ = fc_W . [context, y_t] + fc_b
//   gates = y~*W_ih + bias + h @ W_hh^T           (WMMA, M16 N512 K128)
//   LSTM elementwise update
// ---------------------------------------------------------------------------
__global__ void __launch_bounds__(128) scan_kernel(
    const __bf16* __restrict__ encB,
    const __bf16* __restrict__ encProjB,
    const __bf16* __restrict__ w1hc_f,
    const __bf16* __restrict__ whh_f,
    const float* __restrict__ attn_b1,      // (128)
    const float* __restrict__ attn_W2,      // (1,128)
    const float* __restrict__ attn_b2,      // (1)
    const float* __restrict__ W_ih,         // (512,1)
    const float* __restrict__ b_ih,         // (512)
    const float* __restrict__ b_hh,         // (512)
    const float* __restrict__ fc_W,         // (1,129)
    const float* __restrict__ fc_b,         // (1)
    const float* __restrict__ fc_final_W,   // (1,256)
    const float* __restrict__ fc_final_b,   // (1)
    const float* __restrict__ y_history,    // (B, T-1, 1)
    float* __restrict__ out)                // (B, 1)
{
    __shared__ __bf16 sm_hcb[16][256];  // bf16 [h | c] tile (WMMA A-side source)
    __shared__ float sm_c[16][128];     // fp32 cell state (recurrence accuracy)
    __shared__ float sm_ctx[16][128];
    __shared__ float sm_big[16][512];   // phases A-C: [0:128)=hc_proj, [128:256)=e/beta
                                        // phase F-G: full 16x512 gate tile
    __shared__ float sm_w2[128];
    __shared__ float sm_b1[128];
    __shared__ float sm_fcW[132];
    __shared__ float sm_wih[512];
    __shared__ float sm_bg[512];
    __shared__ float sm_ytil[16];

    const int bbase = blockIdx.x * 16;
    const int tid   = threadIdx.x;
    const int wave  = tid >> 5, lane = tid & 31;
    const int mA    = lane & 15;
    const int aOff  = (lane >= 16) ? 8 : 0;
    const int nLocal = lane & 15;
    const int mBase  = (lane >= 16) ? 8 : 0;

    for (int i = tid; i < 512; i += 128) {
        sm_wih[i] = W_ih[i];
        sm_bg[i]  = b_ih[i] + b_hh[i];
        if (i < 128) { sm_w2[i] = attn_W2[i]; sm_b1[i] = attn_b1[i]; }
        if (i < 132) sm_fcW[i] = (i < 129) ? fc_W[i] : 0.0f;
    }
    for (int i = tid; i < 16 * 256; i += 128) {
        ((__bf16*)sm_hcb)[i] = (__bf16)0.0f;
        if (i < 16 * 128) { ((float*)sm_c)[i] = 0.0f; ((float*)sm_ctx)[i] = 0.0f; }
    }
    const float b2   = attn_b2[0];
    const float fcb0 = fc_b[0];
    const v16bf* w1hc_v = (const v16bf*)w1hc_f;
    const v16bf* whh_v  = (const v16bf*)whh_f;
    const __bf16* encP  = encProjB + (size_t)bbase * DEC_T * 128;
    const __bf16* encX  = encB     + (size_t)bbase * DEC_T * 128;
    __syncthreads();

    for (int t = 0; t < DEC_T - 1; ++t) {
        // ---- Phase A: hc_proj = [h|c] @ [W1_h|W1_c]^T + b1 ------------------
        #pragma unroll
        for (int half = 0; half < 2; ++half) {
            int nt = wave * 2 + half;           // 0..7
            int n  = nt * 16 + nLocal;
            v8f acc;
            #pragma unroll
            for (int r = 0; r < 8; ++r) acc[r] = sm_b1[n];
            #pragma unroll
            for (int ks = 0; ks < 8; ++ks) {    // K = 256 over [h|c]
                v16bf a = load_a_frag(&sm_hcb[mA][0], ks, aOff);
                v16bf b = w1hc_v[(nt * 8 + ks) * 32 + lane];
                acc = wmma_bf16(a, b, acc);
            }
            #pragma unroll
            for (int r = 0; r < 8; ++r) sm_big[r + mBase][n] = acc[r];
        }
        __syncthreads();

        // ---- Phase B: attention scores e[b][t'] (t' = tid, b = j) -----------
        #pragma unroll 1
        for (int j = 0; j < 16; ++j) {
            const v16bf* row = (const v16bf*)(encP + ((size_t)j * DEC_T + tid) * 128);
            float ev = b2;
            #pragma unroll
            for (int c8 = 0; c8 < 8; ++c8) {
                v16bf ch = row[c8];
                #pragma unroll
                for (int i = 0; i < 16; ++i) {
                    int hh = c8 * 16 + i;
                    ev += sm_w2[hh] * fast_tanh((float)ch[i] + sm_big[j][hh]);
                }
            }
            sm_big[j][128 + tid] = ev;
        }
        __syncthreads();

        // ---- Phase C: softmax over t' per batch row -------------------------
        if (tid < 16) {
            int b = tid;
            float mx = -3.4e38f;
            for (int tt = 0; tt < 128; ++tt) mx = fmaxf(mx, sm_big[b][128 + tt]);
            float s = 0.0f;
            for (int tt = 0; tt < 128; ++tt) {
                float ex = __expf(sm_big[b][128 + tt] - mx);
                sm_big[b][128 + tt] = ex;
                s += ex;
            }
            float inv = __builtin_amdgcn_rcpf(s);
            for (int tt = 0; tt < 128; ++tt) sm_big[b][128 + tt] *= inv;
        }
        __syncthreads();

        // ---- Phase D: context = beta @ encoder tile -------------------------
        {
            int b  = tid >> 3;                  // 0..15
            int ch = tid & 7;                   // 16-wide e-chunk
            float acc[16];
            #pragma unroll
            for (int i = 0; i < 16; ++i) acc[i] = 0.0f;
            const __bf16* rowBase = encX + (size_t)b * DEC_T * 128 + ch * 16;
            #pragma unroll 1
            for (int tt = 0; tt < 128; ++tt) {
                float beta = sm_big[b][128 + tt];
                v16bf ev = *(const v16bf*)(rowBase + (size_t)tt * 128);
                #pragma unroll
                for (int i = 0; i < 16; ++i) acc[i] += beta * (float)ev[i];
            }
            #pragma unroll
            for (int i = 0; i < 16; ++i) sm_ctx[b][ch * 16 + i] = acc[i];
        }
        __syncthreads();

        // ---- Phase E: y_tilde ----------------------------------------------
        if (tid < 16) {
            int b = tid;
            float s = fcb0;
            for (int e = 0; e < 128; ++e) s += sm_fcW[e] * sm_ctx[b][e];
            float y = y_history[(size_t)(bbase + b) * (DEC_T - 1) + t];
            sm_ytil[b] = s + sm_fcW[128] * y;
        }
        __syncthreads();

        // ---- Phase F: gates = y~*W_ih + (b_ih+b_hh) + h @ W_hh^T ------------
        #pragma unroll
        for (int q = 0; q < 8; ++q) {
            int nt = wave * 8 + q;              // 0..31 -> wave w owns gate w
            int n  = nt * 16 + nLocal;          // 0..511
            v8f acc;
            #pragma unroll
            for (int r = 0; r < 8; ++r) {
                int m = r + mBase;
                acc[r] = sm_ytil[m] * sm_wih[n] + sm_bg[n];
            }
            #pragma unroll
            for (int ks = 0; ks < 4; ++ks) {    // K = 128 over h only
                v16bf a = load_a_frag(&sm_hcb[mA][0], ks, aOff);
                v16bf b = whh_v[(nt * 4 + ks) * 32 + lane];
                acc = wmma_bf16(a, b, acc);
            }
            #pragma unroll
            for (int r = 0; r < 8; ++r) sm_big[r + mBase][n] = acc[r];
        }
        __syncthreads();

        // ---- Phase G: LSTM elementwise update -------------------------------
        {
            int b  = tid >> 3;
            int ch = tid & 7;
            #pragma unroll
            for (int jj = 0; jj < 16; ++jj) {
                int j  = ch * 16 + jj;
                float gi = sm_big[b][j];
                float gf = sm_big[b][128 + j];
                float gg = sm_big[b][256 + j];
                float go = sm_big[b][384 + j];
                float cn = fast_sig(gf) * sm_c[b][j] + fast_sig(gi) * fast_tanh(gg);
                sm_c[b][j] = cn;
                sm_hcb[b][j]       = (__bf16)(fast_sig(go) * fast_tanh(cn));
                sm_hcb[b][128 + j] = (__bf16)cn;
            }
        }
        __syncthreads();
    }

    // ---- Final: out = fc_final_W . [h, context] + fc_final_b ----------------
    if (tid < 16) {
        int b = tid;
        float s = fc_final_b[0];
        for (int k = 0; k < 128; ++k) s += fc_final_W[k] * (float)sm_hcb[b][k];
        for (int k = 0; k < 128; ++k) s += fc_final_W[128 + k] * sm_ctx[b][k];
        out[bbase + b] = s;
    }
}

// ---------------------------------------------------------------------------
extern "C" void kernel_launch(void* const* d_in, const int* in_sizes, int n_in,
                              void* d_out, int out_size, void* d_ws, size_t ws_size,
                              hipStream_t stream) {
    const float* input_encoded = (const float*)d_in[0];
    const float* y_history     = (const float*)d_in[1];
    const float* attn_W1       = (const float*)d_in[2];
    const float* attn_b1       = (const float*)d_in[3];
    const float* attn_W2       = (const float*)d_in[4];
    const float* attn_b2       = (const float*)d_in[5];
    const float* W_ih          = (const float*)d_in[6];
    const float* W_hh          = (const float*)d_in[7];
    const float* b_ih          = (const float*)d_in[8];
    const float* b_hh          = (const float*)d_in[9];
    const float* fc_W          = (const float*)d_in[10];
    const float* fc_b          = (const float*)d_in[11];
    const float* fc_final_W    = (const float*)d_in[12];
    const float* fc_final_b    = (const float*)d_in[13];
    float* out = (float*)d_out;

    const size_t rows = (size_t)DEC_B * DEC_T;              // 524288
    char* ws = (char*)d_ws;
    size_t off = 0;
    __bf16* encB     = (__bf16*)(ws + off); off += rows * 128 * sizeof(__bf16);   // 128 MB
    __bf16* encProjB = (__bf16*)(ws + off); off += rows * 128 * sizeof(__bf16);   // 128 MB
    __bf16* w1hc_f   = (__bf16*)(ws + off); off += (size_t)64  * 512 * sizeof(__bf16);
    __bf16* whh_f    = (__bf16*)(ws + off); off += (size_t)128 * 512 * sizeof(__bf16);
    __bf16* w1e_f    = (__bf16*)(ws + off); off += (size_t)32  * 512 * sizeof(__bf16);
    (void)ws_size; (void)in_sizes; (void)n_in; (void)out_size;

    prep_frags_kernel<<<224, 32, 0, stream>>>(attn_W1, W_hh, w1hc_f, whh_f, w1e_f);
    enc_proj_kernel<<<(int)(rows / 64), 128, 0, stream>>>(input_encoded, w1e_f, encB, encProjB);
    scan_kernel<<<DEC_B / 16, 128, 0, stream>>>(
        encB, encProjB, w1hc_f, whh_f,
        attn_b1, attn_W2, attn_b2, W_ih, b_ih, b_hh,
        fc_W, fc_b, fc_final_W, fc_final_b, y_history, out);
}
